// SABlockWithRoPE_32203664785989
// MI455X (gfx1250) — compile-verified, matrix operand order checked by reading, not run on hardware
//
#include <hip/hip_runtime.h>
#include <math.h>

// ---------------------------------------------------------------------------
// Types for CDNA5 WMMA (wave32): v_wmma_f32_16x16x32_bf16
// ---------------------------------------------------------------------------
typedef __bf16 bf16_t;
typedef __attribute__((ext_vector_type(16))) __bf16 bf16x16;
typedef __attribute__((ext_vector_type(8)))  __bf16 bf16x8;
typedef __attribute__((ext_vector_type(8)))  float  f32x8;
typedef __attribute__((ext_vector_type(4)))  float  f32x4;

#define NUM_HEADS 16
#define HEAD_DIM  64
#define SEQ_LEN   2048
#define CDIM      1024
#define FDIM      3072
#define BATCH     2
#define ROPE_BASE 10000.0f

__device__ __forceinline__ f32x8 wmma_bf16(bf16x16 a, bf16x16 b, f32x8 c) {
  // 8 args: (neg_a, A, neg_b, B, c_mod, C, reuse_a, reuse_b)
  return __builtin_amdgcn_wmma_f32_16x16x32_bf16(false, a, false, b,
                                                 (short)0, c, false, false);
}

// ---------------------------------------------------------------------------
// Fragment loader (bf16 source, global or LDS).
// A-matrix 16x32 bf16 layout (ISA 7.12.2): lane = half*16 + m
//   element e (0..15): K = (e&7) + half*8 + (e>>3)*16, row M = lane&15.
// B-matrix 32x16 is symmetric (N per lane, same K packing), so the same
// loader works when the source is [N][K] row-major (K contiguous).
// ---------------------------------------------------------------------------
__device__ __forceinline__ bf16x16 load_frag_bf16(const bf16_t* __restrict__ rowp,
                                                  int k0, int half) {
  const bf16_t* p = rowp + k0 + half * 8;
  bf16x8 lo = *(const bf16x8*)(p);
  bf16x8 hi = *(const bf16x8*)(p + 16);
  bf16x16 r;
#pragma unroll
  for (int i = 0; i < 8; ++i) { r[i] = lo[i]; r[8 + i] = hi[i]; }
  return r;
}

// Row reductions over the 16 lanes sharing one C-fragment row.
__device__ __forceinline__ float rowred_max(float v) {
#pragma unroll
  for (int m = 1; m < 16; m <<= 1) v = fmaxf(v, __shfl_xor(v, m, 32));
  return v;
}
__device__ __forceinline__ float rowred_sum(float v) {
#pragma unroll
  for (int m = 1; m < 16; m <<= 1) v += __shfl_xor(v, m, 32);
  return v;
}

// ---------------------------------------------------------------------------
// Kernel 0: streaming fp32 -> bf16 conversion (8 elements / thread).
// ---------------------------------------------------------------------------
__global__ __launch_bounds__(256) void cvt_f32_to_bf16_kernel(
    const float* __restrict__ src, bf16_t* __restrict__ dst, int n) {
  const int i = (blockIdx.x * 256 + threadIdx.x) * 8;
  if (i >= n) return;
  f32x4 a = *(const f32x4*)(src + i);
  f32x4 b = *(const f32x4*)(src + i + 4);
  bf16x8 o;
#pragma unroll
  for (int j = 0; j < 4; ++j) {
    o[j]     = (__bf16)a[j];
    o[4 + j] = (__bf16)b[j];
  }
  *(bf16x8*)(dst + i) = o;
}

// ---------------------------------------------------------------------------
// Kernel 1: qkv = x @ w_qkv^T (bf16 in, f32 acc), RoPE on q/k in the epilogue.
//   q,k stored [b*H+h][l][d]; v stored transposed [b*H+h][d][l].
// Block: 256 thr = 8 waves; block tile 256(M) x 64(N); wave tile 32x64.
// ---------------------------------------------------------------------------
__global__ __launch_bounds__(256) void qkv_rope_kernel(
    const bf16_t* __restrict__ x, const bf16_t* __restrict__ w_qkv,
    bf16_t* __restrict__ q, bf16_t* __restrict__ k, bf16_t* __restrict__ vT) {
  const int lane = threadIdx.x & 31;
  const int wave = threadIdx.x >> 5;
  const int half = lane >> 4;
  const int m0 = blockIdx.x * 256 + wave * 32;  // row in B*L
  const int n0 = blockIdx.y * 64;               // col in 3*C

  f32x8 acc[2][4] = {};
  const bf16_t* arow0 = x + (size_t)(m0 + (lane & 15)) * CDIM;
  const bf16_t* arow1 = x + (size_t)(m0 + 16 + (lane & 15)) * CDIM;
  for (int k0 = 0; k0 < CDIM; k0 += 32) {
    bf16x16 a0 = load_frag_bf16(arow0, k0, half);
    bf16x16 a1 = load_frag_bf16(arow1, k0, half);
#pragma unroll
    for (int j = 0; j < 4; ++j) {
      const bf16_t* brow = w_qkv + (size_t)(n0 + 16 * j + (lane & 15)) * CDIM;
      bf16x16 b = load_frag_bf16(brow, k0, half);
      acc[0][j] = wmma_bf16(a0, b, acc[0][j]);
      acc[1][j] = wmma_bf16(a1, b, acc[1][j]);
    }
  }

  const float lbase = logf(ROPE_BASE) * (1.0f / 64.0f);
#pragma unroll
  for (int t = 0; t < 2; ++t) {
#pragma unroll
    for (int j = 0; j < 4; ++j) {
#pragma unroll
      for (int r = 0; r < 8; ++r) {
        float val = acc[t][j][r];
        float partner = __shfl_xor(val, 1, 32);  // lane n^1 holds f^1
        const int m = m0 + t * 16 + r + half * 8;
        const int f = n0 + 16 * j + (lane & 15);
        const int b   = m >> 11;      // / SEQ_LEN
        const int l   = m & 2047;
        const int s   = f >> 10;      // 0=q 1=k 2=v
        const int rem = f & 1023;
        const int h   = rem >> 6;
        const int d   = rem & 63;
        float outv = val;
        if (s < 2) {
          const int fj = d >> 1;                       // freq index 0..31
          const float ang = (float)l * expf(-(float)fj * lbase);
          float sn, cs;
          sincosf(ang, &sn, &cs);
          outv = (d & 1) ? (partner * sn + val * cs)   // odd:  te*sin + to*cos
                         : (val * cs - partner * sn);  // even: te*cos - to*sin
        }
        const bf16_t ov = (bf16_t)outv;
        const size_t bh = (size_t)(b * NUM_HEADS + h);
        if (s == 0)
          q[(bh * SEQ_LEN + l) * HEAD_DIM + d] = ov;
        else if (s == 1)
          k[(bh * SEQ_LEN + l) * HEAD_DIM + d] = ov;
        else
          vT[(bh * HEAD_DIM + d) * SEQ_LEN + l] = ov;
      }
    }
  }
}

// ---------------------------------------------------------------------------
// Kernel 2: flash attention, online softmax, cooperative KV staging.
// Block = 8 waves = 128 query rows of one (b,h). KV tiles of 32 keys are
// staged in double-buffered LDS by all 256 threads (2 x b128 global loads per
// thread per tile), so K/V global traffic drops 8x vs per-wave streaming.
// Software pipeline: issue tile i+1 loads -> compute tile i from LDS ->
// store tile i+1 regs to LDS -> one barrier per iteration.
// ---------------------------------------------------------------------------
__global__ __launch_bounds__(256) void flash_attn_kernel(
    const bf16_t* __restrict__ q, const bf16_t* __restrict__ k,
    const bf16_t* __restrict__ vT, bf16_t* __restrict__ o) {
  __shared__ __align__(16) bf16_t klds[2][32 * 64];  // keys x d       (4KB x2)
  __shared__ __align__(16) bf16_t vlds[2][64 * 32];  // d x keys       (4KB x2)
  __shared__ __align__(16) bf16_t plds[8][16 * 32];  // per-wave P     (1KB x8)

  const int tid  = threadIdx.x;
  const int lane = tid & 31;
  const int wave = tid >> 5;
  const int half = lane >> 4;
  const int bh = blockIdx.y;                       // b*H + h
  const int q0 = blockIdx.x * 128 + wave * 16;

  const bf16_t* qb = q  + (size_t)bh * SEQ_LEN * HEAD_DIM;
  const bf16_t* kb = k  + (size_t)bh * SEQ_LEN * HEAD_DIM;
  const bf16_t* vb = vT + (size_t)bh * HEAD_DIM * SEQ_LEN;

  const bf16_t* qrow = qb + (size_t)(q0 + (lane & 15)) * HEAD_DIM;
  const bf16x16 qf0 = load_frag_bf16(qrow, 0, half);   // d = 0..31
  const bf16x16 qf1 = load_frag_bf16(qrow, 32, half);  // d = 32..63

  // staging roles: one b128 of K-tile + one b128 of V-tile per thread
  const int krow_s = tid >> 3, kcol_s = (tid & 7) * 8;   // K: 32 x 64
  const int vrow_s = tid >> 2, vcol_s = (tid & 3) * 8;   // V: 64 x 32

  // prologue: stage tile kv0 = 0 into buffer 0
  {
    bf16x8 nk = *(const bf16x8*)(kb + (size_t)krow_s * HEAD_DIM + kcol_s);
    bf16x8 nv = *(const bf16x8*)(vb + (size_t)vrow_s * SEQ_LEN + vcol_s);
    *(bf16x8*)(&klds[0][krow_s * 64 + kcol_s]) = nk;
    *(bf16x8*)(&vlds[0][vrow_s * 32 + vcol_s]) = nv;
  }
  __syncthreads();

  f32x8 acc[4] = {};
  float m_i[8], l_i[8];
#pragma unroll
  for (int r = 0; r < 8; ++r) { m_i[r] = -INFINITY; l_i[r] = 0.0f; }

  const float scale = 0.125f;  // hd^-0.5
  bf16_t* pl = &plds[wave][0];

  for (int kv0 = 0; kv0 < SEQ_LEN; kv0 += 32) {
    const int cur = (kv0 >> 5) & 1;
    const bool have_next = (kv0 + 32) < SEQ_LEN;

    // issue next tile's global loads early (latency hidden under compute)
    bf16x8 nk, nv;
    if (have_next) {
      nk = *(const bf16x8*)(kb + (size_t)(kv0 + 32 + krow_s) * HEAD_DIM + kcol_s);
      nv = *(const bf16x8*)(vb + (size_t)vrow_s * SEQ_LEN + kv0 + 32 + vcol_s);
    }

    // ---- S = Q * K^T  (16 x 32 scores as two 16x16 WMMA tiles) ----
    f32x8 s0 = {}, s1 = {};
    {
      const bf16_t* krow0 = &klds[cur][(lane & 15) * 64];
      const bf16_t* krow1 = &klds[cur][(16 + (lane & 15)) * 64];
      s0 = wmma_bf16(qf0, load_frag_bf16(krow0, 0, half), s0);
      s0 = wmma_bf16(qf1, load_frag_bf16(krow0, 32, half), s0);
      s1 = wmma_bf16(qf0, load_frag_bf16(krow1, 0, half), s1);
      s1 = wmma_bf16(qf1, load_frag_bf16(krow1, 32, half), s1);
    }
    // ---- online softmax over the 32 new keys ----
#pragma unroll
    for (int r = 0; r < 8; ++r) {
      const float a  = s0[r] * scale;
      const float b2 = s1[r] * scale;
      const float mx = rowred_max(fmaxf(a, b2));
      const float m_new = fmaxf(m_i[r], mx);
      const float alpha = expf(m_i[r] - m_new);
      const float pa = expf(a - m_new);
      const float pb = expf(b2 - m_new);
      l_i[r] = l_i[r] * alpha + rowred_sum(pa + pb);
      m_i[r] = m_new;
#pragma unroll
      for (int j = 0; j < 4; ++j) acc[j][r] *= alpha;
      const int mrow = r + half * 8;
      pl[mrow * 32 + (lane & 15)]      = (bf16_t)pa;  // keys kv0..kv0+15
      pl[mrow * 32 + 16 + (lane & 15)] = (bf16_t)pb;  // keys kv0+16..kv0+31
    }
    // ---- transpose P through LDS into A-fragment layout (in-order DS) ----
    const bf16x16 pf = load_frag_bf16(pl + (size_t)(lane & 15) * 32, 0, half);
    // ---- O += P * V  (V staged transposed: row d, keys contiguous) ----
#pragma unroll
    for (int j = 0; j < 4; ++j) {
      const bf16_t* vrow = &vlds[cur][(16 * j + (lane & 15)) * 32];
      acc[j] = wmma_bf16(pf, load_frag_bf16(vrow, 0, half), acc[j]);
    }

    // ---- stage next tile into the other buffer ----
    if (have_next) {
      *(bf16x8*)(&klds[cur ^ 1][krow_s * 64 + kcol_s]) = nk;
      *(bf16x8*)(&vlds[cur ^ 1][vrow_s * 32 + vcol_s]) = nv;
    }
    __syncthreads();
  }

  // ---- epilogue: normalize, store in (B, L, H*hd) layout for out-proj ----
  const int b = bh >> 4;
  const int h = bh & 15;
#pragma unroll
  for (int j = 0; j < 4; ++j) {
#pragma unroll
    for (int r = 0; r < 8; ++r) {
      const float ov = acc[j][r] / l_i[r];
      const size_t row = (size_t)b * SEQ_LEN + q0 + r + half * 8;
      const int col = h * HEAD_DIM + 16 * j + (lane & 15);
      o[row * CDIM + col] = (bf16_t)ov;
    }
  }
}

// ---------------------------------------------------------------------------
// Kernel 3: out = attn_out @ w_out^T + b_out (fp32 output).
// Wave tile 32x64, bf16 operands throughout.
// ---------------------------------------------------------------------------
__global__ __launch_bounds__(256) void out_proj_kernel(
    const bf16_t* __restrict__ a_in, const bf16_t* __restrict__ w_out,
    const float* __restrict__ b_out, float* __restrict__ out) {
  const int lane = threadIdx.x & 31;
  const int wave = threadIdx.x >> 5;
  const int half = lane >> 4;
  const int m0 = blockIdx.x * 256 + wave * 32;
  const int n0 = blockIdx.y * 64;

  f32x8 acc[2][4] = {};
  const bf16_t* arow0 = a_in + (size_t)(m0 + (lane & 15)) * CDIM;
  const bf16_t* arow1 = a_in + (size_t)(m0 + 16 + (lane & 15)) * CDIM;
  for (int k0 = 0; k0 < CDIM; k0 += 32) {
    bf16x16 a0 = load_frag_bf16(arow0, k0, half);
    bf16x16 a1 = load_frag_bf16(arow1, k0, half);
#pragma unroll
    for (int j = 0; j < 4; ++j) {
      const bf16_t* brow = w_out + (size_t)(n0 + 16 * j + (lane & 15)) * CDIM;
      bf16x16 b = load_frag_bf16(brow, k0, half);
      acc[0][j] = wmma_bf16(a0, b, acc[0][j]);
      acc[1][j] = wmma_bf16(a1, b, acc[1][j]);
    }
  }
#pragma unroll
  for (int j = 0; j < 4; ++j) {
    const int n = n0 + 16 * j + (lane & 15);
    const float bias = b_out[n];
#pragma unroll
    for (int t = 0; t < 2; ++t) {
#pragma unroll
      for (int r = 0; r < 8; ++r) {
        const size_t row = (size_t)(m0 + t * 16 + r + half * 8);
        out[row * CDIM + n] = acc[t][j][r] + bias;
      }
    }
  }
}

// ---------------------------------------------------------------------------
extern "C" void kernel_launch(void* const* d_in, const int* in_sizes, int n_in,
                              void* d_out, int out_size, void* d_ws,
                              size_t ws_size, hipStream_t stream) {
  const float* x     = (const float*)d_in[0];
  const float* w_qkv = (const float*)d_in[1];
  const float* w_out = (const float*)d_in[2];
  const float* b_out = (const float*)d_in[3];
  float* out = (float*)d_out;

  const size_t n_qkv  = (size_t)BATCH * NUM_HEADS * SEQ_LEN * HEAD_DIM; // 4Mi
  const size_t n_x    = (size_t)BATCH * SEQ_LEN * CDIM;                 // 4Mi
  const size_t n_wqkv = (size_t)FDIM * CDIM;                            // 3Mi
  const size_t n_wout = (size_t)CDIM * CDIM;                            // 1Mi

  bf16_t* q   = (bf16_t*)d_ws;
  bf16_t* k   = q + n_qkv;
  bf16_t* vT  = k + n_qkv;
  bf16_t* ao  = vT + n_qkv;        // attention output, (B*L, C) bf16
  bf16_t* xb  = ao + n_x;          // bf16 copy of x
  bf16_t* wqb = xb + n_x;          // bf16 copy of w_qkv
  bf16_t* wob = wqb + n_wqkv;      // bf16 copy of w_out

  dim3 blk(256);
  cvt_f32_to_bf16_kernel<<<dim3(n_x / (8 * 256)), blk, 0, stream>>>(x, xb, (int)n_x);
  cvt_f32_to_bf16_kernel<<<dim3(n_wqkv / (8 * 256)), blk, 0, stream>>>(w_qkv, wqb, (int)n_wqkv);
  cvt_f32_to_bf16_kernel<<<dim3(n_wout / (8 * 256)), blk, 0, stream>>>(w_out, wob, (int)n_wout);

  dim3 gA((BATCH * SEQ_LEN) / 256, FDIM / 64);  // (16, 48)
  qkv_rope_kernel<<<gA, blk, 0, stream>>>(xb, wqb, q, k, vT);

  dim3 gB(SEQ_LEN / 128, BATCH * NUM_HEADS);    // (16, 32)
  flash_attn_kernel<<<gB, blk, 0, stream>>>(q, k, vT, ao);

  dim3 gC((BATCH * SEQ_LEN) / 256, CDIM / 64);  // (16, 16)
  out_proj_kernel<<<gC, blk, 0, stream>>>(ao, wob, b_out, out);
}
